// SSD_40931038331348
// MI455X (gfx1250) — matrix operand order repaired and checked by reading, not compiled, not run on hardware
//
#include <hip/hip_runtime.h>

// ---------------------------------------------------------------------------
// SSD300 forward on gfx1250 (MI455X): implicit-GEMM convolutions via
// v_wmma_f32_16x16x32_bf16. All intermediate activations live in bf16 NHWC
// (converted once per value in the conv epilogue), so the WMMA K-loop does
// pure vector loads + WMMA. One wave computes MT*16 out-channels x 16
// positions, reusing the B fragment across MT accumulators.
// ---------------------------------------------------------------------------

typedef __attribute__((ext_vector_type(16))) __bf16         v16bf;
typedef __attribute__((ext_vector_type(8)))  __bf16         v8bf;
typedef __attribute__((ext_vector_type(8)))  float          v8f;
typedef __attribute__((ext_vector_type(8)))  unsigned short ushort8;
typedef unsigned short ushort;

__device__ __forceinline__ ushort f2bf_bits(float f) {
  unsigned u = __builtin_bit_cast(unsigned, f);
  u += 0x7FFFu + ((u >> 16) & 1u);          // round-to-nearest-even
  return (ushort)(u >> 16);
}
__device__ __forceinline__ float bf2f(ushort b) {
  return __builtin_bit_cast(float, (unsigned)b << 16);
}
__device__ __forceinline__ __bf16 bits2bf(ushort b) {
  return __builtin_bit_cast(__bf16, b);
}

// ---------------------------------------------------------------------------
// Implicit-GEMM conv. X: bf16 NHWC. W: bf16 [OCp][K], k=(ky*KW+kx)*Cin+ci.
// Requires Cin%16==0 and K%32==0 (every conv except c1_1).
// ---------------------------------------------------------------------------
struct ConvParams {
  const ushort* X; const ushort* W; const float* bias; void* Y;
  int B, Hi, Wi, Cin, Ho, Wo, OC, OCp, KH, KW, stride, pad, dil, outBS,
      relu, outF32;
};

template <int MT>
__global__ void conv_wmma(ConvParams p) {
  const int lane  = threadIdx.x & 31;
  const long wave = (long)blockIdx.x * (blockDim.x >> 5) + (threadIdx.x >> 5);
  const int N      = p.B * p.Ho * p.Wo;
  const int Ntiles = (N + 15) >> 4;
  const int Mtiles = p.OCp / (16 * MT);
  if (wave >= (long)Ntiles * Mtiles) return;          // wave-uniform exit
  const int nt   = (int)(wave % Ntiles);
  const int mt   = (int)(wave / Ntiles);
  const int l16  = lane & 15;
  const int half = lane >> 4;

  const int col       = nt * 16 + l16;
  const bool colValid = (col < N);
  const int ccol = colValid ? col : 0;
  const int HW   = p.Ho * p.Wo;
  const int b    = ccol / HW;
  const int pos  = ccol - b * HW;
  const int oy   = pos / p.Wo, ox = pos - (pos / p.Wo) * p.Wo;

  const int K = p.KH * p.KW * p.Cin;
  const ushort* Wr[MT];
#pragma unroll
  for (int mi = 0; mi < MT; ++mi)
    Wr[mi] = p.W + (size_t)((mt * MT + mi) * 16 + l16) * K;

  v8f acc[MT];
#pragma unroll
  for (int mi = 0; mi < MT; ++mi)
    acc[mi] = (v8f){0.f, 0.f, 0.f, 0.f, 0.f, 0.f, 0.f, 0.f};

  for (int k0 = 0; k0 < K; k0 += 32) {
    // ---- B fragment: 16 contiguous bf16 (32B aligned vector load) ----
    v16bf bm;
    const int kB0 = k0 + half * 16;
    const int tap = kB0 / p.Cin;                 // constant across 16 ks
    const int ci0 = kB0 - tap * p.Cin;
    const int ky  = tap / p.KW, kx = tap - (tap / p.KW) * p.KW;
    const int iy  = oy * p.stride - p.pad + ky * p.dil;
    const int ix  = ox * p.stride - p.pad + kx * p.dil;
    if (colValid && (unsigned)iy < (unsigned)p.Hi && (unsigned)ix < (unsigned)p.Wi) {
      const ushort* xp = p.X + (((size_t)b * p.Hi + iy) * p.Wi + ix) * p.Cin + ci0;
      bm = *(const v16bf*)xp;
    } else {
#pragma unroll
      for (int e = 0; e < 16; ++e) bm[e] = bits2bf((ushort)0);
    }
    // ---- MT A fragments share this B fragment ----
#pragma unroll
    for (int mi = 0; mi < MT; ++mi) {
      const ushort* wp0 = Wr[mi] + k0 + half * 16;
      v8bf lo = *(const v8bf*)wp0;
      v8bf hi = *(const v8bf*)(wp0 + 8);
      v16bf a = __builtin_shufflevector(lo, hi, 0, 1, 2, 3, 4, 5, 6, 7,
                                        8, 9, 10, 11, 12, 13, 14, 15);
      acc[mi] = __builtin_amdgcn_wmma_f32_16x16x32_bf16(
          false, a, false, bm, (short)0, acc[mi], false, false);
    }
  }

  if (!colValid) return;
#pragma unroll
  for (int mi = 0; mi < MT; ++mi) {
    const int chanBase = (mt * MT + mi) * 16 + half * 8;
    if (p.outF32) {
      float* yp = (float*)p.Y + (size_t)b * p.outBS + (size_t)pos * p.OC;
#pragma unroll
      for (int j = 0; j < 8; ++j) {
        const int c = chanBase + j;
        if (c < p.OC) {
          float v = acc[mi][j] + p.bias[c];
          if (p.relu) v = fmaxf(v, 0.f);
          yp[c] = v;
        }
      }
    } else {
      ushort* yp = (ushort*)p.Y + (size_t)b * p.outBS + (size_t)pos * p.OC;
      if (chanBase + 8 <= p.OC) {
        ushort8 s;
#pragma unroll
        for (int j = 0; j < 8; ++j) {
          float v = acc[mi][j] + p.bias[chanBase + j];
          if (p.relu) v = fmaxf(v, 0.f);
          s[j] = f2bf_bits(v);
        }
        *(ushort8*)(yp + chanBase) = s;
      } else {
#pragma unroll
        for (int j = 0; j < 8; ++j) {
          const int c = chanBase + j;
          if (c < p.OC) {
            float v = acc[mi][j] + p.bias[c];
            if (p.relu) v = fmaxf(v, 0.f);
            yp[c] = f2bf_bits(v);
          }
        }
      }
    }
  }
}

// ---------------------------------------------------------------------------
// GEMM from a materialized bf16 column buffer (conv1_1, Cin=3, Kp=32).
// ---------------------------------------------------------------------------
__global__ void gemm_col_wmma(const ushort* col, const ushort* W,
                              const float* bias, ushort* Y,
                              int N, int Kp, int OC, int OCp,
                              int outBS, int HW, int relu) {
  const int lane  = threadIdx.x & 31;
  const long wave = (long)blockIdx.x * (blockDim.x >> 5) + (threadIdx.x >> 5);
  const int Ntiles = (N + 15) >> 4;
  const int Mtiles = OCp >> 4;
  if (wave >= (long)Ntiles * Mtiles) return;
  const int nt = (int)(wave % Ntiles);
  const int mt = (int)(wave / Ntiles);
  const int l16 = lane & 15, half = lane >> 4;
  const int coln = nt * 16 + l16;
  const bool colValid = (coln < N);
  const int ccol = colValid ? coln : 0;
  const ushort* Wrow = W + (size_t)(mt * 16 + l16) * Kp;
  v8f acc = {0.f, 0.f, 0.f, 0.f, 0.f, 0.f, 0.f, 0.f};
  for (int k0 = 0; k0 < Kp; k0 += 32) {
    const ushort* wp0 = Wrow + k0 + half * 16;
    v8bf lo = *(const v8bf*)wp0;
    v8bf hi = *(const v8bf*)(wp0 + 8);
    v16bf a = __builtin_shufflevector(lo, hi, 0, 1, 2, 3, 4, 5, 6, 7,
                                      8, 9, 10, 11, 12, 13, 14, 15);
    v16bf bm = *(const v16bf*)(col + (size_t)ccol * Kp + k0 + half * 16);
    acc = __builtin_amdgcn_wmma_f32_16x16x32_bf16(false, a, false, bm,
                                                  (short)0, acc, false, false);
  }
  if (!colValid) return;
  const int b = ccol / HW, pos = ccol - (ccol / HW) * HW;
  const int chanBase = mt * 16 + half * 8;
  ushort* yp = Y + (size_t)b * outBS + (size_t)pos * OC;
  ushort8 s;
#pragma unroll
  for (int j = 0; j < 8; ++j) {
    float v = acc[j] + bias[chanBase + j];
    if (relu) v = fmaxf(v, 0.f);
    s[j] = f2bf_bits(v);
  }
  *(ushort8*)(yp + chanBase) = s;
}

// ---------------------------------------------------------------------------
// Helpers
// ---------------------------------------------------------------------------
__global__ void nchw2nhwc_bf16(const float* X, ushort* Y, int B, int C, int H,
                               int W) {
  long idx = (long)blockIdx.x * blockDim.x + threadIdx.x;
  long total = (long)B * C * H * W;
  if (idx >= total) return;
  int x = idx % W; long t = idx / W;
  int y = t % H; t /= H;
  int c = t % C; int b = (int)(t / C);
  Y[(((long)b * H + y) * W + x) * C + c] = f2bf_bits(X[idx]);
}

// conv1_1: X bf16 NHWC [4,300,300,3] -> col[n][32], k=(ky*3+kx)*3+ci, 0 pad
__global__ void im2col_c1(const ushort* X, ushort* col) {
  int n = blockIdx.x * blockDim.x + threadIdx.x;
  if (n >= 4 * 300 * 300) return;
  int b = n / 90000, pos = n % 90000, oy = pos / 300, ox = pos % 300;
  ushort* out = col + (size_t)n * 32;
#pragma unroll
  for (int k = 0; k < 32; ++k) {
    ushort v = 0;
    if (k < 27) {
      int tap = k / 3, ci = k % 3, ky = tap / 3, kx = tap % 3;
      int iy = oy - 1 + ky, ix = ox - 1 + kx;
      if (iy >= 0 && iy < 300 && ix >= 0 && ix < 300)
        v = X[(((long)b * 300 + iy) * 300 + ix) * 3 + ci];
    }
    out[k] = v;
  }
}

// OIHW f32 -> bf16 [OCp][Kp], k=(ky*KW+kx)*Cin+ci (zeros for padded oc/k)
__global__ void pack_w(const float* w, ushort* out,
                       int OC, int Cin, int KH, int KW, int OCp, int Kp) {
  long idx = (long)blockIdx.x * blockDim.x + threadIdx.x;
  long total = (long)OCp * Kp;
  if (idx >= total) return;
  int oc = (int)(idx / Kp), k = (int)(idx % Kp);
  float v = 0.f;
  int Kv = KH * KW * Cin;
  if (oc < OC && k < Kv) {
    int tap = k / Cin, ci = k % Cin, ky = tap / KW, kx = tap % KW;
    v = w[(((long)oc * Cin + ci) * KH + ky) * KW + kx];
  }
  out[idx] = f2bf_bits(v);
}

__global__ void maxpool_bf16(const ushort* X, ushort* Y, int B, int Hi, int Wi,
                             int C, int Ho, int Wo, int k, int s, int padLo) {
  long idx = (long)blockIdx.x * blockDim.x + threadIdx.x;
  long total = (long)B * Ho * Wo * C;
  if (idx >= total) return;
  int c = idx % C; long t = idx / C;
  int ox = t % Wo; t /= Wo;
  int oy = t % Ho; int b = (int)(t / Ho);
  float m = -3.4e38f;
  for (int dy = 0; dy < k; ++dy) {
    int iy = oy * s - padLo + dy;
    if (iy < 0 || iy >= Hi) continue;
    for (int dx = 0; dx < k; ++dx) {
      int ix = ox * s - padLo + dx;
      if (ix < 0 || ix >= Wi) continue;
      float v = bf2f(X[(((long)b * Hi + iy) * Wi + ix) * C + c]);
      m = v > m ? v : m;
    }
  }
  Y[idx] = f2bf_bits(m);   // exact: m is already a bf16 value
}

// per-position L2 norm over channels, then scale by rescale[c]
__global__ void l2norm_rescale(const ushort* X, const float* rescale, ushort* Y,
                               int C) {
  int row = blockIdx.x;
  __shared__ float red[256];
  float s = 0.f;
  for (int c = threadIdx.x; c < C; c += blockDim.x) {
    float v = bf2f(X[(long)row * C + c]);
    s += v * v;
  }
  red[threadIdx.x] = s;
  __syncthreads();
  for (int off = 128; off > 0; off >>= 1) {
    if ((int)threadIdx.x < off) red[threadIdx.x] += red[threadIdx.x + off];
    __syncthreads();
  }
  float inv = rsqrtf(red[0]);
  for (int c = threadIdx.x; c < C; c += blockDim.x)
    Y[(long)row * C + c] = f2bf_bits(bf2f(X[(long)row * C + c]) * inv * rescale[c]);
}

// ---------------------------------------------------------------------------
// Host orchestration
// ---------------------------------------------------------------------------
namespace {

struct PSpec { int OC, Cin, K; };
enum {
  P_C10_1, P_C10_2, P_C11_1, P_C11_2, P_C1_1, P_C1_2, P_C2_1, P_C2_2,
  P_C3_1, P_C3_2, P_C3_3, P_C4_1, P_C4_2, P_C4_3, P_C5_1, P_C5_2, P_C5_3,
  P_C8_1, P_C8_2, P_C9_1, P_C9_2,
  P_CL10, P_CL11, P_CL4, P_CL7, P_CL8, P_CL9,
  P_CONV6, P_CONV7,
  P_LOC10, P_LOC11, P_LOC4, P_LOC7, P_LOC8, P_LOC9,
  NPARAMS
};
// JAX pytree order: dict keys sorted lexicographically
static const PSpec kSpecs[NPARAMS] = {
  {128, 256, 1}, {256, 128, 3}, {128, 256, 1}, {256, 128, 3},       // c10_1..c11_2
  {64, 3, 3},    {64, 64, 3},                                       // c1_1, c1_2
  {128, 64, 3},  {128, 128, 3},                                     // c2_1, c2_2
  {256, 128, 3}, {256, 256, 3}, {256, 256, 3},                      // c3_1..c3_3
  {512, 256, 3}, {512, 512, 3}, {512, 512, 3},                      // c4_1..c4_3
  {512, 512, 3}, {512, 512, 3}, {512, 512, 3},                      // c5_1..c5_3
  {256, 1024, 1}, {512, 256, 3},                                    // c8_1, c8_2
  {128, 512, 1},  {256, 128, 3},                                    // c9_1, c9_2
  {84, 256, 3}, {84, 256, 3}, {84, 512, 3},                         // cl10, cl11, cl4
  {126, 1024, 3}, {126, 512, 3}, {126, 256, 3},                     // cl7, cl8, cl9
  {1024, 512, 3}, {1024, 1024, 1},                                  // conv6, conv7
  {16, 256, 3}, {16, 256, 3}, {16, 512, 3},                         // loc10, loc11, loc4
  {24, 1024, 3}, {24, 512, 3}, {24, 256, 3},                        // loc7, loc8, loc9
};

inline size_t alignUp(size_t x) { return (x + 255) & ~size_t(255); }

}  // namespace

extern "C" void kernel_launch(void* const* d_in, const int* in_sizes, int n_in,
                              void* d_out, int out_size, void* d_ws, size_t ws_size,
                              hipStream_t stream) {
  (void)in_sizes; (void)n_in; (void)out_size; (void)ws_size;
  const int BATCH = 4;

  const float* images = (const float*)d_in[0];
  const float* Wp[NPARAMS];
  const float* Bp[NPARAMS];
  for (int p = 0; p < NPARAMS; ++p) {
    Wp[p] = (const float*)d_in[1 + 2 * p];
    Bp[p] = (const float*)d_in[2 + 2 * p];
  }
  const float* rescale = (const float*)d_in[1 + 2 * NPARAMS];

  // ---- workspace layout ----
  char* ws = (char*)d_ws;
  size_t cur = 0;
  ushort* wbuf[NPARAMS];
  int OCp[NPARAMS], Kp[NPARAMS];
  for (int p = 0; p < NPARAMS; ++p) {
    const PSpec& s = kSpecs[p];
    OCp[p] = (s.OC + 15) & ~15;
    int K = s.K * s.K * s.Cin;
    Kp[p] = (K + 31) & ~31;
    wbuf[p] = (ushort*)(ws + cur);
    cur = alignUp(cur + (size_t)OCp[p] * Kp[p] * 2);
  }
  const size_t ACT_BYTES = (size_t)BATCH * 300 * 300 * 64 * 2;  // 46 MB (bf16)
  ushort* ACT0 = (ushort*)(ws + cur); cur = alignUp(cur + ACT_BYTES);
  ushort* ACT1 = (ushort*)(ws + cur); cur = alignUp(cur + ACT_BYTES);
  ushort* P4  = (ushort*)(ws + cur); cur = alignUp(cur + (size_t)BATCH * 38 * 38 * 512 * 2);
  ushort* P7  = (ushort*)(ws + cur); cur = alignUp(cur + (size_t)BATCH * 19 * 19 * 1024 * 2);
  ushort* P8  = (ushort*)(ws + cur); cur = alignUp(cur + (size_t)BATCH * 10 * 10 * 512 * 2);
  ushort* P9  = (ushort*)(ws + cur); cur = alignUp(cur + (size_t)BATCH * 5 * 5 * 256 * 2);
  ushort* P10 = (ushort*)(ws + cur); cur = alignUp(cur + (size_t)BATCH * 3 * 3 * 256 * 2);
  ushort* P11 = (ushort*)(ws + cur); cur = alignUp(cur + (size_t)BATCH * 1 * 1 * 256 * 2);
  ushort* COL = (ushort*)(ws + cur);
  cur = alignUp(cur + (size_t)BATCH * 300 * 300 * 32 * 2);

  // ---- pack all weights to bf16 ----
  for (int p = 0; p < NPARAMS; ++p) {
    long tot = (long)OCp[p] * Kp[p];
    pack_w<<<(unsigned)((tot + 255) / 256), 256, 0, stream>>>(
        Wp[p], wbuf[p], kSpecs[p].OC, kSpecs[p].Cin, kSpecs[p].K, kSpecs[p].K,
        OCp[p], Kp[p]);
  }

  // ---- input NCHW f32 -> NHWC bf16 ----
  {
    long tot = (long)BATCH * 3 * 300 * 300;
    nchw2nhwc_bf16<<<(unsigned)((tot + 255) / 256), 256, 0, stream>>>(
        images, ACT0, BATCH, 3, 300, 300);
  }

  // ---- conv1_1 via im2col + column GEMM (Cin=3) ----
  im2col_c1<<<(BATCH * 300 * 300 + 255) / 256, 256, 0, stream>>>(ACT0, COL);
  {
    int N = BATCH * 300 * 300;
    long waves = (long)((N + 15) / 16) * (OCp[P_C1_1] / 16);
    gemm_col_wmma<<<(unsigned)((waves + 7) / 8), 256, 0, stream>>>(
        COL, wbuf[P_C1_1], Bp[P_C1_1], ACT1,
        N, Kp[P_C1_1], 64, OCp[P_C1_1], 300 * 300 * 64, 300 * 300, 1);
  }

  auto conv = [&](int pidx, const ushort* X, void* Y, int Hi, int Wi,
                  int Ho, int Wo, int stride, int pad, int dil,
                  int outBS, int relu, int outF32) {
    const PSpec& s = kSpecs[pidx];
    ConvParams cp;
    cp.X = X; cp.W = wbuf[pidx]; cp.bias = Bp[pidx]; cp.Y = Y;
    cp.B = BATCH; cp.Hi = Hi; cp.Wi = Wi; cp.Cin = s.Cin;
    cp.Ho = Ho; cp.Wo = Wo; cp.OC = s.OC; cp.OCp = OCp[pidx];
    cp.KH = s.K; cp.KW = s.K; cp.stride = stride; cp.pad = pad; cp.dil = dil;
    cp.outBS = outBS; cp.relu = relu; cp.outF32 = outF32;
    int MT = (cp.OCp % 64 == 0) ? 4 : ((cp.OCp % 32 == 0) ? 2 : 1);
    int N = BATCH * Ho * Wo;
    long waves = (long)((N + 15) / 16) * (cp.OCp / (16 * MT));
    unsigned blocks = (unsigned)((waves + 7) / 8);
    if (MT == 4)      conv_wmma<4><<<blocks, 256, 0, stream>>>(cp);
    else if (MT == 2) conv_wmma<2><<<blocks, 256, 0, stream>>>(cp);
    else              conv_wmma<1><<<blocks, 256, 0, stream>>>(cp);
  };
  auto pool = [&](const ushort* X, ushort* Y, int Hi, int Wi, int C,
                  int Ho, int Wo, int k, int s, int padLo) {
    long tot = (long)BATCH * Ho * Wo * C;
    maxpool_bf16<<<(unsigned)((tot + 255) / 256), 256, 0, stream>>>(
        X, Y, BATCH, Hi, Wi, C, Ho, Wo, k, s, padLo);
  };

  // ---- VGG backbone ----
  conv(P_C1_2, ACT1, ACT0, 300, 300, 300, 300, 1, 1, 1, 300 * 300 * 64, 1, 0);
  pool(ACT0, ACT1, 300, 300, 64, 150, 150, 2, 2, 0);
  conv(P_C2_1, ACT1, ACT0, 150, 150, 150, 150, 1, 1, 1, 150 * 150 * 128, 1, 0);
  conv(P_C2_2, ACT0, ACT1, 150, 150, 150, 150, 1, 1, 1, 150 * 150 * 128, 1, 0);
  pool(ACT1, ACT0, 150, 150, 128, 75, 75, 2, 2, 0);
  conv(P_C3_1, ACT0, ACT1, 75, 75, 75, 75, 1, 1, 1, 75 * 75 * 256, 1, 0);
  conv(P_C3_2, ACT1, ACT0, 75, 75, 75, 75, 1, 1, 1, 75 * 75 * 256, 1, 0);
  conv(P_C3_3, ACT0, ACT1, 75, 75, 75, 75, 1, 1, 1, 75 * 75 * 256, 1, 0);
  pool(ACT1, ACT0, 75, 75, 256, 38, 38, 2, 2, 0);   // ceil mode: pad-high only
  conv(P_C4_1, ACT0, ACT1, 38, 38, 38, 38, 1, 1, 1, 38 * 38 * 512, 1, 0);
  conv(P_C4_2, ACT1, ACT0, 38, 38, 38, 38, 1, 1, 1, 38 * 38 * 512, 1, 0);
  conv(P_C4_3, ACT0, ACT1, 38, 38, 38, 38, 1, 1, 1, 38 * 38 * 512, 1, 0);
  // conv4_3 -> L2 normalized + rescaled feature
  l2norm_rescale<<<BATCH * 38 * 38, 256, 0, stream>>>(ACT1, rescale, P4, 512);
  pool(ACT1, ACT0, 38, 38, 512, 19, 19, 2, 2, 0);
  conv(P_C5_1, ACT0, ACT1, 19, 19, 19, 19, 1, 1, 1, 19 * 19 * 512, 1, 0);
  conv(P_C5_2, ACT1, ACT0, 19, 19, 19, 19, 1, 1, 1, 19 * 19 * 512, 1, 0);
  conv(P_C5_3, ACT0, ACT1, 19, 19, 19, 19, 1, 1, 1, 19 * 19 * 512, 1, 0);
  pool(ACT1, ACT0, 19, 19, 512, 19, 19, 3, 1, 1);   // k3 s1 p1
  conv(P_CONV6, ACT0, ACT1, 19, 19, 19, 19, 1, 6, 6, 19 * 19 * 1024, 1, 0);
  conv(P_CONV7, ACT1, P7, 19, 19, 19, 19, 1, 0, 1, 19 * 19 * 1024, 1, 0);

  // ---- extra feature layers ----
  conv(P_C8_1, P7, ACT0, 19, 19, 19, 19, 1, 0, 1, 19 * 19 * 256, 1, 0);
  conv(P_C8_2, ACT0, P8, 19, 19, 10, 10, 2, 1, 1, 10 * 10 * 512, 1, 0);
  conv(P_C9_1, P8, ACT0, 10, 10, 10, 10, 1, 0, 1, 10 * 10 * 128, 1, 0);
  conv(P_C9_2, ACT0, P9, 10, 10, 5, 5, 2, 1, 1, 5 * 5 * 256, 1, 0);
  conv(P_C10_1, P9, ACT0, 5, 5, 5, 5, 1, 0, 1, 5 * 5 * 128, 1, 0);
  conv(P_C10_2, ACT0, P10, 5, 5, 3, 3, 1, 0, 1, 3 * 3 * 256, 1, 0);
  conv(P_C11_1, P10, ACT0, 3, 3, 3, 3, 1, 0, 1, 3 * 3 * 128, 1, 0);
  conv(P_C11_2, ACT0, P11, 3, 3, 1, 1, 1, 0, 1, 1 * 1 * 256, 1, 0);

  // ---- detection heads: NHWC conv output == transpose+reshape of reference;
  //      write f32 directly into d_out at per-feature box offsets ----
  float* locBase = (float*)d_out;                             // (4, 8732, 4)
  float* clsBase = (float*)d_out + (size_t)BATCH * 8732 * 4;  // (4, 8732, 21)
  struct Head { int pLoc, pCls; const ushort* feat; int H; int off; };
  const Head heads[6] = {
    {P_LOC4,  P_CL4,  P4,  38, 0},
    {P_LOC7,  P_CL7,  P7,  19, 5776},
    {P_LOC8,  P_CL8,  P8,  10, 7942},
    {P_LOC9,  P_CL9,  P9,   5, 8542},
    {P_LOC10, P_CL10, P10,  3, 8692},
    {P_LOC11, P_CL11, P11,  1, 8728},
  };
  for (int h = 0; h < 6; ++h) {
    const Head& hd = heads[h];
    conv(hd.pLoc, hd.feat, locBase + (size_t)hd.off * 4,
         hd.H, hd.H, hd.H, hd.H, 1, 1, 1, 8732 * 4, 0, 1);
    conv(hd.pCls, hd.feat, clsBase + (size_t)hd.off * 21,
         hd.H, hd.H, hd.H, hd.H, 1, 1, 1, 8732 * 21, 0, 1);
  }
}
